// CustomMultiheadAttention_29386166239354
// MI455X (gfx1250) — matrix operand order repaired and checked by reading, not compiled, not run on hardware
//
#include <hip/hip_runtime.h>
#include <hip/hip_bf16.h>
#include <stdint.h>

typedef __bf16 bf16;
typedef __bf16 bf16x8  __attribute__((ext_vector_type(8)));
typedef __bf16 v16bf   __attribute__((ext_vector_type(16)));
typedef float  floatx8 __attribute__((ext_vector_type(8)));
typedef float  floatx4 __attribute__((ext_vector_type(4)));

#define EMBED 1024
#define SEQ   2048
#define NH    16
#define HD    64
#define BATCH 4

#define BM 128
#define BN 128
#define BK 32
#define LDA 40   // padded LDS row stride (elements); 80B rows keep 16B alignment

__device__ __forceinline__ bf16x8 ld8(const bf16* p) { return *(const bf16x8*)p; }

__device__ __forceinline__ v16bf combine16(bf16x8 a, bf16x8 b) {
  return __builtin_shufflevector(a, b, 0,1,2,3,4,5,6,7,8,9,10,11,12,13,14,15);
}

__device__ __forceinline__ floatx8 wmma_bf16(v16bf a, v16bf b, floatx8 c) {
  return __builtin_amdgcn_wmma_f32_16x16x32_bf16(false, a, false, b, (short)0, c,
                                                 false, false);
}

__device__ __forceinline__ float rowmax16(float v) {
#pragma unroll
  for (int m = 1; m < 16; m <<= 1) v = fmaxf(v, __shfl_xor(v, m, 16));
  return v;
}
__device__ __forceinline__ float rowsum16(float v) {
#pragma unroll
  for (int m = 1; m < 16; m <<= 1) v += __shfl_xor(v, m, 16);
  return v;
}

// gfx1250 async global->LDS 16B copy (tracked by ASYNCcnt).
// LDS destination offset = low 32 bits of the flat shared pointer (LDS aperture).
__device__ __forceinline__ void async_cp16(bf16* lds_dst, const bf16* gsrc) {
  uint32_t dst = (uint32_t)(uintptr_t)lds_dst;
  asm volatile("global_load_async_to_lds_b128 %0, %1, off"
               :: "v"(dst), "v"(gsrc) : "memory");
}
__device__ __forceinline__ void wait_async0() {
  asm volatile("s_wait_asynccnt 0x0" ::: "memory");
}

// ---------------------------------------------------------------------------
// GEMM: out = X(f32, MxK) @ W(f32, NxK)^T + bias, output bf16.
// mode 0: store [B,H,S,D]  (Q, K)
// mode 1: store [B,H,D,S]  (V transposed, for contiguous WMMA-B loads later)
// ---------------------------------------------------------------------------
__global__ __launch_bounds__(256) void qkv_proj_kernel(
    const float* __restrict__ X, const float* __restrict__ W,
    const float* __restrict__ bias, bf16* __restrict__ out, int mode) {
  __shared__ bf16 As[BM * LDA];
  __shared__ bf16 Bs[BN * LDA];

  const int t  = threadIdx.x;
  const int ln = t & 31, w = t >> 5;
  const int wr = w >> 1, wc = w & 1;           // 4x2 wave grid -> 32x64 per wave
  const int lo = ln & 15, hi = ln >> 4;
  const int m0 = blockIdx.y * BM;
  const int n0 = blockIdx.x * BN;

  floatx8 acc[2][4];
#pragma unroll
  for (int i = 0; i < 2; ++i)
#pragma unroll
    for (int j = 0; j < 4; ++j) acc[i][j] = (floatx8)0.0f;

  for (int kk = 0; kk < EMBED; kk += BK) {
#pragma unroll
    for (int i = 0; i < 4; ++i) {            // 1024 float4 tile loads, 4/thread
      int f = t + i * 256;
      int row = f >> 3;
      int c4 = (f & 7) << 2;
      floatx4 xv = *(const floatx4*)(X + (size_t)(m0 + row) * EMBED + kk + c4);
      floatx4 wv = *(const floatx4*)(W + (size_t)(n0 + row) * EMBED + kk + c4);
      bf16* da = &As[row * LDA + c4];
      bf16* db = &Bs[row * LDA + c4];
#pragma unroll
      for (int j = 0; j < 4; ++j) { da[j] = (bf16)xv[j]; db[j] = (bf16)wv[j]; }
    }
    __syncthreads();

    v16bf afrag[2];
#pragma unroll
    for (int mf = 0; mf < 2; ++mf) {
      int arow = wr * 32 + mf * 16 + lo;
      afrag[mf] = combine16(ld8(&As[arow * LDA + hi * 8]),
                            ld8(&As[arow * LDA + hi * 8 + 16]));
    }
#pragma unroll
    for (int nf = 0; nf < 4; ++nf) {
      int brow = wc * 64 + nf * 16 + lo;
      v16bf bfrag = combine16(ld8(&Bs[brow * LDA + hi * 8]),
                              ld8(&Bs[brow * LDA + hi * 8 + 16]));
#pragma unroll
      for (int mf = 0; mf < 2; ++mf)
        acc[mf][nf] = wmma_bf16(afrag[mf], bfrag, acc[mf][nf]);
    }
    __syncthreads();
  }

#pragma unroll
  for (int mf = 0; mf < 2; ++mf) {
#pragma unroll
    for (int nf = 0; nf < 4; ++nf) {
      int n = n0 + wc * 64 + nf * 16 + lo;
      float bv = bias[n];
      int h = n >> 6, d = n & 63;
      int mbase = m0 + wr * 32 + mf * 16 + 8 * hi;
      int bidx = mbase >> 11;                 // / SEQ
      int sbase = mbase & (SEQ - 1);
      if (mode == 0) {
#pragma unroll
        for (int r = 0; r < 8; ++r) {
          out[(((size_t)bidx * NH + h) * SEQ + (sbase + r)) * HD + d] =
              (bf16)(acc[mf][nf][r] + bv);
        }
      } else {
        bf16x8 vvec;
#pragma unroll
        for (int r = 0; r < 8; ++r) vvec[r] = (bf16)(acc[mf][nf][r] + bv);
        *(bf16x8*)&out[(((size_t)bidx * NH + h) * HD + d) * SEQ + sbase] = vvec;
      }
    }
  }
}

// ---------------------------------------------------------------------------
// Flash attention (causal). One wave = 16 query rows. KV tiles (32 wide) are
// staged block-wide in LDS via async global->LDS copies with double buffering;
// all 8 waves consume the same staged tiles (8x less K/V traffic).
// Q,K: [B,H,S,64] bf16.  VT: [B,H,64,S] bf16.  Ctx out: [B,S,1024] bf16.
// ---------------------------------------------------------------------------
__global__ __launch_bounds__(256) void flash_attn_kernel(
    const bf16* __restrict__ Q, const bf16* __restrict__ K,
    const bf16* __restrict__ VT, bf16* __restrict__ Ctx) {
  __shared__ bf16 Ks[2][32 * 64];            // [kv-row][d]
  __shared__ bf16 Vs[2][64 * 32];            // [d][kv-row]  (from VT)
  __shared__ bf16 pbuf[8][16 * 32];          // per-wave P staging (C->A relayout)

  const int t = threadIdx.x, ln = t & 31, w = t >> 5;
  const int lo = ln & 15, hi = ln >> 4;
  const int qpb = SEQ / 128;
  const int qblk = blockIdx.x % qpb;
  const int bh = blockIdx.x / qpb;
  const int b = bh / NH, h = bh % NH;
  const int qs = qblk * 128 + w * 16;        // this wave's first query row (s)

  const bf16* qbase = Q + (size_t)bh * SEQ * HD;
  const bf16* kbase = K + (size_t)bh * SEQ * HD;
  const bf16* vbase = VT + (size_t)bh * HD * SEQ;
  bf16* pb = &pbuf[w][0];

  // per-thread 16B copy slots for the staged tiles
  const int krow = t >> 3, kchunk = (t & 7) * 8;   // K: 32 rows x 64 d
  const int vrow = t >> 2, vchunk = (t & 3) * 8;   // V: 64 d-rows x 32 s

  auto issue_tile = [&](int kv, int buf) {
    async_cp16(&Ks[buf][krow * 64 + kchunk],
               kbase + (size_t)(kv + krow) * HD + kchunk);
    async_cp16(&Vs[buf][vrow * 32 + vchunk],
               vbase + (size_t)vrow * SEQ + kv + vchunk);
  };

  // Q fragments (A-layout): row = qs+lo, K chunks (hi*8 +{0..7,16..23}) over D
  const bf16* qrow = qbase + (size_t)(qs + lo) * HD;
  v16bf aq0 = combine16(ld8(qrow + hi * 8), ld8(qrow + hi * 8 + 16));
  v16bf aq1 = combine16(ld8(qrow + 32 + hi * 8), ld8(qrow + 32 + hi * 8 + 16));

  floatx8 O[4];
#pragma unroll
  for (int dt = 0; dt < 4; ++dt) O[dt] = (floatx8)0.0f;
  float m_r[8], l_r[8];
#pragma unroll
  for (int r = 0; r < 8; ++r) { m_r[r] = -1e30f; l_r[r] = 0.0f; }

  const float SC = 0.125f * 1.44269504f;     // 1/sqrt(64) * log2(e)
  const int blk_end = qblk * 128 + 128;      // block-uniform causal bound
  const int my_end = qs + 16;                // this wave's causal bound

  issue_tile(0, 0);
  int buf = 0;
  for (int kv = 0; kv < blk_end; kv += 32, buf ^= 1) {
    wait_async0();                           // my copies into `buf` landed
    __syncthreads();                         // everyone's copies landed; everyone
                                             // done reading buf^1 from last iter
    if (kv + 32 < blk_end) issue_tile(kv + 32, buf ^ 1);

    if (kv < my_end) {
      // K^T B-frags from LDS: col n=lo -> K row (ct*16+lo), d = dbase+hi*8+{0..7,16..23}
      const bf16* k0 = &Ks[buf][(size_t)lo * 64];
      const bf16* k1 = &Ks[buf][(size_t)(16 + lo) * 64];
      v16bf bk00 = combine16(ld8(k0 + hi * 8), ld8(k0 + hi * 8 + 16));
      v16bf bk01 = combine16(ld8(k0 + 32 + hi * 8), ld8(k0 + 32 + hi * 8 + 16));
      v16bf bk10 = combine16(ld8(k1 + hi * 8), ld8(k1 + hi * 8 + 16));
      v16bf bk11 = combine16(ld8(k1 + 32 + hi * 8), ld8(k1 + 32 + hi * 8 + 16));

      floatx8 s0 = (floatx8)0.0f, s1 = (floatx8)0.0f;
      s0 = wmma_bf16(aq0, bk00, s0);
      s0 = wmma_bf16(aq1, bk01, s0);
      s1 = wmma_bf16(aq0, bk10, s1);
      s1 = wmma_bf16(aq1, bk11, s1);

#pragma unroll
      for (int r = 0; r < 8; ++r) {
        int row = qs + r + 8 * hi;
        s0[r] = (kv + lo <= row) ? s0[r] * SC : -1e30f;
        s1[r] = (kv + 16 + lo <= row) ? s1[r] * SC : -1e30f;
        float cand = rowmax16(fmaxf(s0[r], s1[r]));
        float mn = fmaxf(m_r[r], cand);
        float rescale = exp2f(m_r[r] - mn);
        m_r[r] = mn;
        float p0 = exp2f(s0[r] - mn);
        float p1 = exp2f(s1[r] - mn);
        l_r[r] = l_r[r] * rescale + rowsum16(p0 + p1);
#pragma unroll
        for (int dt = 0; dt < 4; ++dt) O[dt][r] *= rescale;
        pb[(r + 8 * hi) * 32 + lo] = (bf16)p0;
        pb[(r + 8 * hi) * 32 + 16 + lo] = (bf16)p1;
      }
      __builtin_amdgcn_wave_barrier();
      // P A-fragment from LDS (row = lo, K chunks hi*8 +{0..7,16..23})
      v16bf ap = combine16(ld8(pb + lo * 32 + hi * 8),
                           ld8(pb + lo * 32 + hi * 8 + 16));
      __builtin_amdgcn_wave_barrier();

#pragma unroll
      for (int dt = 0; dt < 4; ++dt) {
        // V B-frag: lane -> d-row (dt*16+lo), s = hi*8 + {0..7,16..23}
        const bf16* vp = &Vs[buf][(size_t)(dt * 16 + lo) * 32 + hi * 8];
        v16bf bvf = combine16(ld8(vp), ld8(vp + 16));
        O[dt] = wmma_bf16(ap, bvf, O[dt]);
      }
    }
  }

#pragma unroll
  for (int r = 0; r < 8; ++r) {
    float inv = 1.0f / l_r[r];
    int sidx = qs + r + 8 * hi;
    size_t base = ((size_t)b * SEQ + sidx) * EMBED + h * HD;
#pragma unroll
    for (int dt = 0; dt < 4; ++dt)
      Ctx[base + dt * 16 + lo] = (bf16)(O[dt][r] * inv);
  }
}

// ---------------------------------------------------------------------------
// Output projection: out(f32) = Ctx(bf16, Mx1024) @ Wo(f32,1024x1024)^T + bo
// ---------------------------------------------------------------------------
__global__ __launch_bounds__(256) void out_proj_kernel(
    const bf16* __restrict__ Ctx, const float* __restrict__ W,
    const float* __restrict__ bias, float* __restrict__ Out) {
  __shared__ bf16 As[BM * LDA];
  __shared__ bf16 Bs[BN * LDA];

  const int t = threadIdx.x;
  const int ln = t & 31, w = t >> 5;
  const int wr = w >> 1, wc = w & 1;
  const int lo = ln & 15, hi = ln >> 4;
  const int m0 = blockIdx.y * BM;
  const int n0 = blockIdx.x * BN;

  floatx8 acc[2][4];
#pragma unroll
  for (int i = 0; i < 2; ++i)
#pragma unroll
    for (int j = 0; j < 4; ++j) acc[i][j] = (floatx8)0.0f;

  for (int kk = 0; kk < EMBED; kk += BK) {
    {  // Ctx tile: already bf16, 32B per thread
      int row = t >> 1;
      int off = (t & 1) * 16;
      const bf16* src = Ctx + (size_t)(m0 + row) * EMBED + kk + off;
      *(bf16x8*)&As[row * LDA + off] = ld8(src);
      *(bf16x8*)&As[row * LDA + off + 8] = ld8(src + 8);
    }
#pragma unroll
    for (int i = 0; i < 4; ++i) {  // Wo tile: f32 -> bf16
      int f = t + i * 256;
      int row = f >> 3;
      int c4 = (f & 7) << 2;
      floatx4 wv = *(const floatx4*)(W + (size_t)(n0 + row) * EMBED + kk + c4);
      bf16* db = &Bs[row * LDA + c4];
#pragma unroll
      for (int j = 0; j < 4; ++j) db[j] = (bf16)wv[j];
    }
    __syncthreads();

    v16bf afrag[2];
#pragma unroll
    for (int mf = 0; mf < 2; ++mf) {
      int arow = wr * 32 + mf * 16 + lo;
      afrag[mf] = combine16(ld8(&As[arow * LDA + hi * 8]),
                            ld8(&As[arow * LDA + hi * 8 + 16]));
    }
#pragma unroll
    for (int nf = 0; nf < 4; ++nf) {
      int brow = wc * 64 + nf * 16 + lo;
      v16bf bfrag = combine16(ld8(&Bs[brow * LDA + hi * 8]),
                              ld8(&Bs[brow * LDA + hi * 8 + 16]));
#pragma unroll
      for (int mf = 0; mf < 2; ++mf)
        acc[mf][nf] = wmma_bf16(afrag[mf], bfrag, acc[mf][nf]);
    }
    __syncthreads();
  }

#pragma unroll
  for (int mf = 0; mf < 2; ++mf) {
#pragma unroll
    for (int nf = 0; nf < 4; ++nf) {
      int n = n0 + wc * 64 + nf * 16 + lo;
      float bv = bias[n];
      int mbase = m0 + wr * 32 + mf * 16 + 8 * hi;
#pragma unroll
      for (int r = 0; r < 8; ++r)
        Out[(size_t)(mbase + r) * EMBED + n] = acc[mf][nf][r] + bv;
    }
  }
}

extern "C" void kernel_launch(void* const* d_in, const int* in_sizes, int n_in,
                              void* d_out, int out_size, void* d_ws,
                              size_t ws_size, hipStream_t stream) {
  (void)in_sizes; (void)n_in; (void)out_size; (void)ws_size;
  const float* q  = (const float*)d_in[0];
  const float* k  = (const float*)d_in[1];
  const float* v  = (const float*)d_in[2];
  // d_in[3] = attn_mask (causal) — implemented analytically in-kernel
  const float* Wq = (const float*)d_in[4];
  const float* bq = (const float*)d_in[5];
  const float* Wk = (const float*)d_in[6];
  const float* bk = (const float*)d_in[7];
  const float* Wv = (const float*)d_in[8];
  const float* bv = (const float*)d_in[9];
  const float* Wo = (const float*)d_in[10];
  const float* bo = (const float*)d_in[11];

  char* ws = (char*)d_ws;
  const size_t SEG = (size_t)16 << 20;       // 16 MB per buffer
  bf16* qt  = (bf16*)(ws + 0 * SEG);         // [B,H,S,64]
  bf16* kt  = (bf16*)(ws + 1 * SEG);         // [B,H,S,64]
  bf16* vt  = (bf16*)(ws + 2 * SEG);         // [B,H,64,S]
  bf16* ctx = (bf16*)(ws + 3 * SEG);         // [B,S,1024]

  dim3 block(256);
  dim3 gGemm(EMBED / BN, (BATCH * SEQ) / BM);              // (8, 64)
  qkv_proj_kernel<<<gGemm, block, 0, stream>>>(q, Wq, bq, qt, 0);
  qkv_proj_kernel<<<gGemm, block, 0, stream>>>(k, Wk, bk, kt, 0);
  qkv_proj_kernel<<<gGemm, block, 0, stream>>>(v, Wv, bv, vt, 1);

  dim3 gAttn(BATCH * NH * (SEQ / 128));                    // 1024 blocks
  flash_attn_kernel<<<gAttn, block, 0, stream>>>(qt, kt, vt, ctx);

  out_proj_kernel<<<gGemm, block, 0, stream>>>(ctx, Wo, bo, (float*)d_out);
}
